// SummaRunnerSentenceExtractor_68298569941444
// MI455X (gfx1250) — matrix operand order repaired and checked by reading, not compiled, not run on hardware
//
#include <hip/hip_runtime.h>
#include <hip/hip_bf16.h>

#define B_    256
#define S_    200
#define IN_   1024
#define SENT_ 100
#define DOC_  200
#define NPOS_ 50
#define NSEG_ 4
#define PSZ_  50
#define SSZ_  50

#define M_TOT (B_ * S_)      // 51200 rows of the big GEMM
#define NT    7              // ceil(SENT/16) N-tiles (112 padded cols)

#define KSTEPS      (IN_ / 32)        // 32 k-steps of K=32
#define KC_STEPS    8                 // k-steps staged per LDS chunk (256 K)
#define N_CHUNKS    (KSTEPS / KC_STEPS)
#define CHUNK_U4    (KC_STEPS * NT * 64)   // 3584 uint4 = 56 KB
#define PACKW_U4    (KSTEPS  * NT * 64)    // 14336 uint4 = 224 KB

typedef __attribute__((ext_vector_type(16))) __bf16 v16bf;
typedef __attribute__((ext_vector_type(8)))  float  v8f;

union bfrag_u { v16bf v; uint4 u[2]; };

static __device__ __forceinline__ __bf16 f2bf(float f) { return (__bf16)f; }

static __device__ __forceinline__ float wave_reduce_sum(float v) {
  #pragma unroll
  for (int o = 16; o > 0; o >>= 1) v += __shfl_xor(v, o, 32);
  return v;
}

// ---------------------------------------------------------------------------
// Kernel 0: pack W_sent [IN, SENT] into bf16 WMMA-B fragments.
// Fragment (ks,t) stored as TWO 16B-per-lane planes (16B lane stride =>
// 16 lanes x 16B = 256B = full 64-bank width => conflict-free ds_load_b128).
// ---------------------------------------------------------------------------
__global__ __launch_bounds__(256) void pack_W_kernel(
    const float* __restrict__ W, uint4* __restrict__ packedW)
{
  const int wid  = (blockIdx.x * blockDim.x + threadIdx.x) >> 5;
  const int lane = threadIdx.x & 31;
  if (wid >= KSTEPS * NT) return;
  const int ks    = wid / NT;
  const int t     = wid % NT;
  const int k0    = ks * 32;
  const int khalf = lane >> 4;
  const int n     = t * 16 + (lane & 15);
  const bool nv   = (n < SENT_);

  bfrag_u frag;
  #pragma unroll
  for (int e = 0; e < 16; ++e) {
    const int k = k0 + khalf * 16 + e;     // contiguous k = 16*khalf + e (same order as A)
    frag.v[e] = f2bf(nv ? W[(size_t)k * SENT_ + n] : 0.0f);
  }
  packedW[(size_t)(wid * 2 + 0) * 32 + lane] = frag.u[0];
  packedW[(size_t)(wid * 2 + 1) * 32 + lane] = frag.u[1];
}

// ---------------------------------------------------------------------------
// Kernel 1: states[B*S, SENT] = relu(encoder_output[B*S, IN] @ W_sent + b)
// bf16 WMMA, f32 accumulate. 8 waves/block; each wave: 16 rows x 112 cols.
// Packed W staged into LDS via CDNA5 async DMA (global_load_async_to_lds).
// Per k-step: 14 ds_load_b128 into 7 distinct fragment buffers, a sched
// fence, then 7 WMMAs back-to-back.
// ---------------------------------------------------------------------------
__global__ __launch_bounds__(256) void gemm_states_kernel(
    const float* __restrict__ A,      // [M_TOT, IN]
    const uint4* __restrict__ packedW,
    const float* __restrict__ bvec,   // [SENT]
    float* __restrict__ states)       // [M_TOT, SENT]
{
  __shared__ uint4 Wlds[CHUNK_U4];    // 56 KB

  const int wave  = threadIdx.x >> 5;
  const int lane  = threadIdx.x & 31;
  const int mtile = blockIdx.x * 8 + wave;
  const int row0  = mtile * 16;
  const int arow  = row0 + (lane & 15);
  const int khalf = lane >> 4;          // 0 or 1 (lane group)

  v8f acc[NT] = {};

  const float* Arow = A + (size_t)arow * IN_;
  const unsigned ldsbase = (unsigned)(uintptr_t)(&Wlds[0]);

  for (int kc = 0; kc < N_CHUNKS; ++kc) {
    // ---- stage 56 KB of packed W into LDS via async DMA (ASYNCcnt) ----
    __syncthreads();
    const uint4* src = packedW + (size_t)kc * CHUNK_U4;
    #pragma unroll
    for (int i = 0; i < CHUNK_U4 / 256; ++i) {
      const int idx = i * 256 + threadIdx.x;
      const unsigned            loff  = ldsbase + (unsigned)idx * 16u;
      const unsigned long long  gaddr = (unsigned long long)(uintptr_t)(src + idx);
      asm volatile("global_load_async_to_lds_b128 %0, %1, off"
                   :: "v"(loff), "v"(gaddr) : "memory");
    }
    asm volatile("s_wait_asynccnt 0" ::: "memory");
    __syncthreads();

    #pragma unroll
    for (int ks = 0; ks < KC_STEPS; ++ks) {
      const int k0 = kc * (KC_STEPS * 32) + ks * 32;
      if (k0 + 32 < IN_) __builtin_prefetch(Arow + k0 + 32, 0, 1);

      // A fragment: 16 contiguous f32 at k0 + khalf*16 (matches packed B).
      const float4 a0 = *(const float4*)(Arow + k0 + khalf * 16);
      const float4 a1 = *(const float4*)(Arow + k0 + khalf * 16 + 4);
      const float4 a2 = *(const float4*)(Arow + k0 + khalf * 16 + 8);
      const float4 a3 = *(const float4*)(Arow + k0 + khalf * 16 + 12);
      v16bf afrag;
      afrag[0]  = f2bf(a0.x); afrag[1]  = f2bf(a0.y);
      afrag[2]  = f2bf(a0.z); afrag[3]  = f2bf(a0.w);
      afrag[4]  = f2bf(a1.x); afrag[5]  = f2bf(a1.y);
      afrag[6]  = f2bf(a1.z); afrag[7]  = f2bf(a1.w);
      afrag[8]  = f2bf(a2.x); afrag[9]  = f2bf(a2.y);
      afrag[10] = f2bf(a2.z); afrag[11] = f2bf(a2.w);
      afrag[12] = f2bf(a3.x); afrag[13] = f2bf(a3.y);
      afrag[14] = f2bf(a3.z); afrag[15] = f2bf(a3.w);

      // Issue all B-fragment loads (distinct registers per tile) ...
      bfrag_u bf[NT];
      #pragma unroll
      for (int t = 0; t < NT; ++t) {
        const int fbase = ((ks * NT + t) * 2) * 32 + lane;
        bf[t].u[0] = Wlds[fbase];
        bf[t].u[1] = Wlds[fbase + 32];
      }
      // ... fence so the scheduler cannot sink them into the WMMA stream ...
      __builtin_amdgcn_sched_barrier(0);
      // ... then stream 7 WMMAs back-to-back.
      #pragma unroll
      for (int t = 0; t < NT; ++t) {
        acc[t] = __builtin_amdgcn_wmma_f32_16x16x32_bf16(
            false, afrag, false, bf[t].v, (short)0, acc[t], false, false);
      }
    }
  }

  // Epilogue: bias + relu + store. C layout: VGPR r -> M=r+(lane>=16?8:0).
  const int ncol = lane & 15;
  const int mofs = (lane >> 4) * 8;
  #pragma unroll
  for (int t = 0; t < NT; ++t) {
    const int n = t * 16 + ncol;
    if (n >= SENT_) continue;
    const float bv = bvec[n];
    #pragma unroll
    for (int r = 0; r < 8; ++r) {
      const int   m   = row0 + r + mofs;
      const float val = acc[t][r] + bv;
      states[(size_t)m * SENT_ + n] = val > 0.0f ? val : 0.0f;
    }
  }
}

// ---------------------------------------------------------------------------
// Kernel 2a: avg[b,h] = sum_s states[b,s,h] / length[b]
// ---------------------------------------------------------------------------
__global__ __launch_bounds__(128) void avg_kernel(
    const float* __restrict__ states, const int* __restrict__ length,
    float* __restrict__ avg)
{
  const int b = blockIdx.x;
  const int h = threadIdx.x;
  if (h >= SENT_) return;
  const float* p = states + (size_t)b * S_ * SENT_ + h;
  float s = 0.0f;
  for (int t = 0; t < S_; ++t) s += p[(size_t)t * SENT_];
  avg[b * SENT_ + h] = s / (float)length[b];
}

// ---------------------------------------------------------------------------
// Kernel 2b: doc_rep[b] = tanh(avg[b] @ W_doc1 + b1) @ W_doc2 + b2
// ---------------------------------------------------------------------------
__global__ __launch_bounds__(256) void docrep_kernel(
    const float* __restrict__ avg,
    const float* __restrict__ W1, const float* __restrict__ b1,
    const float* __restrict__ W2, const float* __restrict__ b2,
    float* __restrict__ doc)
{
  __shared__ float av[SENT_];
  __shared__ float hid[DOC_];
  const int b = blockIdx.x;
  for (int h = threadIdx.x; h < SENT_; h += blockDim.x)
    av[h] = avg[b * SENT_ + h];
  __syncthreads();
  for (int d = threadIdx.x; d < DOC_; d += blockDim.x) {
    float s = b1[d];
    for (int h = 0; h < SENT_; ++h) s += av[h] * W1[h * DOC_ + d];
    hid[d] = tanhf(s);
  }
  __syncthreads();
  for (int h = threadIdx.x; h < SENT_; h += blockDim.x) {
    float s = b2[h];
    for (int d = 0; d < DOC_; ++d) s += hid[d] * W2[d * SENT_ + h];
    doc[b * SENT_ + h] = s;
  }
}

// ---------------------------------------------------------------------------
// Kernel 2c: base[b,s] = content + salience + pos_logit + seg_logit + bias
// one wave32 per (b,s); lane-strided dots + shuffle reduction.
// ---------------------------------------------------------------------------
__global__ __launch_bounds__(256) void base_kernel(
    const float* __restrict__ states, const float* __restrict__ doc,
    const float* __restrict__ wc,     const float* __restrict__ bc,
    const float* __restrict__ pos_emb, const float* __restrict__ w_pos,
    const float* __restrict__ seg_emb, const float* __restrict__ w_seg,
    const float* __restrict__ bias,    const int* __restrict__ length,
    float* __restrict__ base)
{
  const int wave = threadIdx.x >> 5;
  const int lane = threadIdx.x & 31;
  const int idx  = blockIdx.x * 8 + wave;      // flattened (b*S + s)
  if (idx >= M_TOT) return;
  const int b = idx / S_;
  const int s = idx % S_;

  const float* st = states + (size_t)idx * SENT_;
  const float* dr = doc + b * SENT_;

  float acc = 0.0f;
  for (int h = lane; h < SENT_; h += 32) {
    const float v = st[h];
    acc += v * (wc[h] + dr[h]);     // content-partial + salience-partial
  }

  const int ap  = s + 1;
  const int apc = ap > NPOS_ ? NPOS_ : ap;
  for (int j = lane; j < PSZ_; j += 32) acc += pos_emb[apc * PSZ_ + j] * w_pos[j];

  const float lenf  = (float)length[b];
  const float chunk = roundf(lenf / (float)NSEG_);
  int rel = (int)ceilf((float)ap / chunk);
  rel = rel < 0 ? 0 : (rel > NSEG_ ? NSEG_ : rel);
  for (int j = lane; j < SSZ_; j += 32) acc += seg_emb[rel * SSZ_ + j] * w_seg[j];

  const float tot = wave_reduce_sum(acc);
  if (lane == 0) base[idx] = tot + bc[0] + bias[0];
}

// ---------------------------------------------------------------------------
// Kernel 3: sequential novelty scan. One workgroup per batch element;
// W_sim staged once in LDS (40 KB << 320 KB/WGP), summary lives in LDS.
// ---------------------------------------------------------------------------
__global__ __launch_bounds__(128) void scan_kernel(
    const float* __restrict__ states, const float* __restrict__ base,
    const float* __restrict__ W_sim,  float* __restrict__ out)
{
  __shared__ float Ws[SENT_ * SENT_];
  __shared__ float summ[SENT_];
  __shared__ float tsum[SENT_];
  __shared__ float red[4];
  __shared__ float gsig;

  const int b = blockIdx.x;
  for (int i = threadIdx.x; i < SENT_ * SENT_; i += blockDim.x) Ws[i] = W_sim[i];
  for (int i = threadIdx.x; i < SENT_; i += blockDim.x) summ[i] = 0.0f;
  __syncthreads();

  const int h    = threadIdx.x;
  const int lane = threadIdx.x & 31;
  const int wv   = threadIdx.x >> 5;
  const float* stb = states + (size_t)b * S_ * SENT_;

  for (int t = 0; t < S_; ++t) {
    if (h < SENT_) tsum[h] = tanhf(summ[h]);
    __syncthreads();

    float part = 0.0f;
    if (h < SENT_) {
      const float* wr = Ws + h * SENT_;
      float m = 0.0f;
      #pragma unroll 4
      for (int k = 0; k < SENT_; ++k) m += wr[k] * tsum[k];
      part = stb[(size_t)t * SENT_ + h] * m;
    }
    part = wave_reduce_sum(part);
    if (lane == 0) red[wv] = part;
    __syncthreads();

    if (threadIdx.x == 0) {
      const float novelty = -(red[0] + red[1] + red[2] + red[3]);
      const float logit   = base[b * S_ + t] + novelty;
      out[(size_t)b * S_ + t] = logit;
      gsig = 1.0f / (1.0f + __expf(-logit));
    }
    __syncthreads();

    if (h < SENT_) summ[h] += stb[(size_t)t * SENT_ + h] * gsig;
    __syncthreads();
  }
}

// ---------------------------------------------------------------------------
extern "C" void kernel_launch(void* const* d_in, const int* in_sizes, int n_in,
                              void* d_out, int out_size, void* d_ws, size_t ws_size,
                              hipStream_t stream) {
  // setup_inputs() order:
  // 0 sentence_emb (unused)  1 length  2 encoder_output  3 encoder_state (unused)
  // 4 W_sent  5 b_sent  6 w_content  7 b_content  8 W_doc1  9 b_doc1
  // 10 W_doc2 11 b_doc2 12 W_sim 13 bias 14 pos_emb 15 w_pos 16 seg_emb 17 w_seg
  const int*   length    = (const int*)  d_in[1];
  const float* enc_out   = (const float*)d_in[2];
  const float* W_sent    = (const float*)d_in[4];
  const float* b_sent    = (const float*)d_in[5];
  const float* w_content = (const float*)d_in[6];
  const float* b_content = (const float*)d_in[7];
  const float* W_doc1    = (const float*)d_in[8];
  const float* b_doc1    = (const float*)d_in[9];
  const float* W_doc2    = (const float*)d_in[10];
  const float* b_doc2    = (const float*)d_in[11];
  const float* W_sim     = (const float*)d_in[12];
  const float* bias      = (const float*)d_in[13];
  const float* pos_emb   = (const float*)d_in[14];
  const float* w_pos     = (const float*)d_in[15];
  const float* seg_emb   = (const float*)d_in[16];
  const float* w_seg     = (const float*)d_in[17];

  float* states = (float*)d_ws;                       // [M_TOT, SENT]
  float* avg    = states + (size_t)M_TOT * SENT_;     // [B, SENT]
  float* doc    = avg    + (size_t)B_ * SENT_;        // [B, SENT]
  float* base   = doc    + (size_t)B_ * SENT_;        // [B, S]
  uint4* packW  = (uint4*)(base + (size_t)M_TOT);     // 224 KB packed W (16B-aligned)
  float* out    = (float*)d_out;                      // [B, S]

  pack_W_kernel<<<(KSTEPS * NT * 32 + 255) / 256, 256, 0, stream>>>(W_sent, packW);
  gemm_states_kernel<<<M_TOT / (8 * 16), 256, 0, stream>>>(enc_out, packW, b_sent, states);
  avg_kernel<<<B_, 128, 0, stream>>>(states, length, avg);
  docrep_kernel<<<B_, 256, 0, stream>>>(avg, W_doc1, b_doc1, W_doc2, b_doc2, doc);
  base_kernel<<<M_TOT / 8, 256, 0, stream>>>(states, doc, w_content, b_content,
                                             pos_emb, w_pos, seg_emb, w_seg,
                                             bias, length, base);
  scan_kernel<<<B_, 128, 0, stream>>>(states, base, W_sim, out);
}
